// RecurrentUNnet_d7_44762149159168
// MI455X (gfx1250) — matrix operand order repaired and checked by reading, not compiled, not run on hardware
//
#include <hip/hip_runtime.h>

// ---------------------------------------------------------------------------
// Types matching CDNA5 WMMA operand layouts (wave32)
// ---------------------------------------------------------------------------
typedef __attribute__((ext_vector_type(16))) __bf16 v16bf;
typedef __attribute__((ext_vector_type(8)))  float  v8f;
typedef __attribute__((ext_vector_type(4)))  unsigned int v4u;
typedef __attribute__((ext_vector_type(8)))  unsigned int v8u;
typedef __attribute__((ext_vector_type(8)))  int  v8i;
typedef __attribute__((ext_vector_type(4)))  int  v4i;

union AU { v16bf v; unsigned short s[16]; v4u q[2]; };

static __device__ __forceinline__ unsigned short f2bf(float f) {
  unsigned u = __float_as_uint(f);
  unsigned r = u + 0x7FFFu + ((u >> 16) & 1u);   // round-to-nearest-even
  return (unsigned short)(r >> 16);
}
static __device__ __forceinline__ float bf2f(unsigned short h) {
  return __uint_as_float(((unsigned)h) << 16);
}
static __device__ __forceinline__ float hsig(float x) {
  return fminf(fmaxf(0.2f * x + 0.5f, 0.0f), 1.0f);
}

// ---------------------------------------------------------------------------
// CDNA5 async data-movement (CDNA5 ISA).
// Preference: builtin TDM -> builtin async-to-LDS -> inline-asm TDM
// (the asm path bypasses clang builtins entirely; D# per section 8).
// ---------------------------------------------------------------------------
#if defined(__has_builtin)
#if __has_builtin(__builtin_amdgcn_tensor_load_to_lds)
#define HAVE_TDM 1
#endif
#if __has_builtin(__builtin_amdgcn_global_load_async_to_lds_b128)
#if __has_builtin(__builtin_amdgcn_s_wait_asynccnt)
#define HAVE_ASYNC 1
#endif
#endif
#endif

#ifdef HAVE_TDM
static __device__ __forceinline__ void tdm_load_1d(unsigned lds_off,
                                                   const void* gsrc,
                                                   unsigned bytes) {
  unsigned long long ga = (unsigned long long)(uintptr_t)gsrc;
  unsigned elems = bytes >> 3;              // data_size=3 -> 8-byte elements
  v4u g0;
  g0[0] = 1u;                               // count=1, user descriptor
  g0[1] = lds_off;                          // lds_addr (bytes)
  g0[2] = (unsigned)(ga & 0xFFFFFFFFu);     // global_addr[31:0]
  g0[3] = (unsigned)((ga >> 32) & 0x1FFFFFFu) | (2u << 30); // [56:32] | type=2
  v8i g1;
  g1[0] = (3 << 16);                        // wg_mask=0, data_size=3 (8B)
  g1[1] = (int)((elems & 0xFFFFu) << 16);   // tensor_dim0[15:0]
  g1[2] = (int)((elems >> 16) & 0xFFFFu) | (1 << 16); // dim0 hi | tensor_dim1=1
  g1[3] = (int)((elems & 0xFFFFu) << 16);   // tile_dim0 = elems
  g1[4] = 1;                                // tile_dim1=1, tile_dim2=0
  g1[5] = (int)elems;                       // tensor_dim0_stride lo
  g1[6] = 0;
  g1[7] = 0;
  v4i z4 = (v4i)0;
#if __clang_major__ >= 23
  v8i z8 = (v8i)0;
  __builtin_amdgcn_tensor_load_to_lds(g0, g1, z4, z4, z8, 0);
#else
  __builtin_amdgcn_tensor_load_to_lds(g0, g1, z4, z4, 0);
#endif
}
#endif

#if !defined(HAVE_TDM) && !defined(HAVE_ASYNC)
// Inline-asm TDM: 2-SGPR-group form (VADDR2/VADDR3 NULL, tensors up to 2D).
// Group0 = 4 SGPRs, Group1 = 8 SGPRs; "s" constraints yield s[a:b] tuples.
static __device__ __forceinline__ void tdm_load_1d_asm(unsigned lds_off,
                                                       const void* gsrc,
                                                       unsigned bytes) {
  unsigned long long ga = (unsigned long long)(uintptr_t)gsrc;
  unsigned elems = bytes >> 3;              // data_size=3 -> 8-byte elements
  v4u g0;
  g0[0] = 1u;                               // count=1, user descriptor
  g0[1] = lds_off;                          // lds_addr (bytes)
  g0[2] = (unsigned)(ga & 0xFFFFFFFFu);     // global_addr[31:0]
  g0[3] = (unsigned)((ga >> 32) & 0x1FFFFFFu) | (2u << 30); // [56:32] | type=2
  v8u g1;
  g1[0] = (3u << 16);                       // wg_mask=0, data_size=3 (8B)
  g1[1] = (elems & 0xFFFFu) << 16;          // tensor_dim0[15:0]
  g1[2] = ((elems >> 16) & 0xFFFFu) | (1u << 16); // dim0 hi | tensor_dim1=1
  g1[3] = (elems & 0xFFFFu) << 16;          // tile_dim0 = elems
  g1[4] = 1u;                               // tile_dim1=1, tile_dim2=0
  g1[5] = elems;                            // tensor_dim0_stride lo
  g1[6] = 0u;
  g1[7] = 0u;
  asm volatile("tensor_load_to_lds %0, %1" :: "s"(g0), "s"(g1) : "memory");
}
#endif

// Stage NT*1024 bytes of repacked weights into LDS.
template <int NT>
static __device__ __forceinline__ void stage_b(const unsigned short* src,
                                               unsigned short* dst,
                                               int tid, int wave) {
#if defined(HAVE_TDM)
  (void)tid;
  if (wave == 0) tdm_load_1d((unsigned)(uintptr_t)dst, src, NT * 1024u);
#elif defined(HAVE_ASYNC)
  (void)wave;
  for (int i = tid; i < NT * 64; i += 128)
    __builtin_amdgcn_global_load_async_to_lds_b128(
        (__attribute__((address_space(1))) void*)(((const v4u*)src) + i),
        (__attribute__((address_space(3))) void*)(((v4u*)dst) + i), 0, 0);
#else
  (void)tid;
  if (wave == 0) tdm_load_1d_asm((unsigned)(uintptr_t)dst, src, NT * 1024u);
#endif
}

static __device__ __forceinline__ void stage_wait() {
#if defined(HAVE_TDM)
  __builtin_amdgcn_s_wait_tensorcnt(0);
#elif defined(HAVE_ASYNC)
  __builtin_amdgcn_s_wait_asynccnt(0);
#else
  asm volatile("s_wait_tensorcnt 0x0" ::: "memory");
#endif
}

// ---------------------------------------------------------------------------
// Weight repack: f32 [K,N] -> bf16 WMMA-B tiles grouped so that one
// (n-group, k-chunk) tile of NT sub-tiles is contiguous (NT*1024 bytes):
// idx = (((ng*nKt + kt)*NT + sub)*32 + lane)*16 + j
// ---------------------------------------------------------------------------
__global__ __launch_bounds__(256) void repack_weights_kernel(
    const float* __restrict__ w, unsigned short* __restrict__ wrep,
    int K, int N, int nKtiles, int NT, int nNgroups) {
  long long idx = (long long)blockIdx.x * blockDim.x + threadIdx.x;
  long long total = (long long)nNgroups * nKtiles * NT * 512;
  if (idx >= total) return;
  int j    = (int)(idx & 15);
  int lane = (int)((idx >> 4) & 31);
  long long r = idx >> 9;
  int sub  = (int)(r % NT); r /= NT;
  int kt   = (int)(r % nKtiles);
  int ng   = (int)(r / nKtiles);
  int k = (kt << 5) + ((j >> 3) << 4) + ((lane >> 4) << 3) + (j & 7);
  int n = ng * (NT << 4) + (sub << 4) + (lane & 15);
  unsigned short val = 0;
  if (k < K && n < N) val = f2bf(w[(size_t)k * N + n]);
  wrep[idx] = val;
}

// ---------------------------------------------------------------------------
// Implicit-GEMM conv (KHxKH, SAME, NHWC) via v_wmma_f32_16x16x32_bf16.
// Block = 4 waves; wave w owns 16 pixels, all waves share an NT*16-wide
// n-strip double-buffered through LDS via TDM / async-to-LDS.
// blockIdx.z batches independent timesteps (input conv). One A gather
// feeds NT WMMAs. accumulate=1 adds into out (recurrent conv(h,U) pass).
// ---------------------------------------------------------------------------
template <int NT>
__global__ __launch_bounds__(128) void conv_wmma_kernel2(
    const unsigned short* __restrict__ in,    // [H*W, Cin] bf16 (+ t stride)
    const unsigned short* __restrict__ wrep,  // repacked bf16 weights
    const float* __restrict__ bias,           // [N] or null
    float* __restrict__ out,                  // [H*W, N] f32 (+ t stride)
    int H, int W, int Cin, int N, int KH, int nKtiles, int accumulate,
    size_t tin_stride, size_t tout_stride) {
  __shared__ __align__(16) unsigned short bsh[2][NT * 512];

  const int tid  = threadIdx.x;
  const int wave = tid >> 5;
  const int lane = tid & 31;
  const int half = lane >> 4;
  const int l15  = lane & 15;
  const int m0   = (blockIdx.x * 4 + wave) << 4;
  const int n0g  = blockIdx.y * (NT << 4);
  const int K    = KH * KH * Cin;
  const int am   = m0 + l15;       // A layout: lane%16 selects M row
  const int ay   = am / W;
  const int ax   = am - ay * W;
  const int khalf = KH >> 1;

  const unsigned short* inz = in + (size_t)blockIdx.z * tin_stride;
  float* outz = out + (size_t)blockIdx.z * tout_stride;

  // base of this block's contiguous weight strip
  const unsigned short* wsrc =
      wrep + (size_t)blockIdx.y * nKtiles * (NT * 512);

  v8f acc[NT];
#pragma unroll
  for (int t = 0; t < NT; ++t) {
    int col = n0g + (t << 4) + l15;
    if (accumulate) {
#pragma unroll
      for (int i = 0; i < 8; ++i) {
        int row = m0 + (half << 3) + i;
        acc[t][i] = (col < N) ? outz[(size_t)row * N + col] : 0.0f;
      }
    } else {
      float bv = (bias != nullptr && col < N) ? bias[col] : 0.0f;
#pragma unroll
      for (int i = 0; i < 8; ++i) acc[t][i] = bv;
    }
  }

  // ---- stage chunk 0 ----
  stage_b<NT>(wsrc, &bsh[0][0], tid, wave);
  stage_wait();
  __syncthreads();

  for (int kt = 0; kt < nKtiles; ++kt) {
    const int k0 = kt << 5;
    const int cur = kt & 1;

    // ---- stage next chunk into the other buffer (overlapped) ----
    if (kt + 1 < nKtiles)
      stage_b<NT>(wsrc + (size_t)(kt + 1) * (NT * 512), &bsh[cur ^ 1][0], tid,
                  wave);

    // ---- A operand: one gather reused by NT WMMAs ----
    AU a;
    if ((Cin & 31) == 0) {
      // 32-wide K chunk lies inside a single (kh,kw) patch -> contiguous ci
      const int p   = k0 / Cin;
      const int cib = k0 - p * Cin;
      const int pr  = p / KH;
      const int dy  = pr - khalf;
      const int dx  = (p - pr * KH) - khalf;
      const int yy = ay + dy, xx = ax + dx;
      const bool inb = (yy >= 0) && (yy < H) && (xx >= 0) && (xx < W);
      v4u z4 = (v4u)0u;
      a.q[0] = z4; a.q[1] = z4;
      if (inb) {
        const unsigned short* ap =
            inz + ((size_t)(yy * W + xx)) * Cin + cib + (half << 3);
        const v4u* aq = (const v4u*)ap;
        a.q[0] = aq[0];   // K = half*8 .. half*8+7
        a.q[1] = aq[2];   // K = 16+half*8 .. 16+half*8+7
      }
    } else {
      // generic (Cin==1 layers): scalar gather with K/bounds padding
#pragma unroll
      for (int j = 0; j < 16; ++j) {
        int v = j >> 1;
        int kk = (v < 4) ? ((v << 1) + (j & 1) + (half << 3))
                         : (16 + ((v - 4) << 1) + (j & 1) + (half << 3));
        int k = k0 + kk;
        unsigned short aval = 0;
        if (k < K) {
          int p  = k / Cin;
          int ci = k - p * Cin;
          int pr = p / KH;
          int yy = ay + pr - khalf;
          int xx = ax + (p - pr * KH) - khalf;
          if (yy >= 0 && yy < H && xx >= 0 && xx < W)
            aval = inz[((size_t)(yy * W + xx)) * Cin + ci];
        }
        a.s[j] = aval;
      }
    }

    // ---- NT WMMAs off the staged B tiles (ds_load_b128 x2 each) ----
#pragma unroll
    for (int t = 0; t < NT; ++t) {
      AU b;
      const v4u* bq = (const v4u*)&bsh[cur][(t << 9) + (lane << 4)];
      b.q[0] = bq[0];
      b.q[1] = bq[1];
      acc[t] = __builtin_amdgcn_wmma_f32_16x16x32_bf16(
          false, a.v, false, b.v, (short)0, acc[t], false, false);
    }

    stage_wait();
    __syncthreads();
  }

#pragma unroll
  for (int t = 0; t < NT; ++t) {
    int col = n0g + (t << 4) + l15;
    if (col < N) {
#pragma unroll
      for (int i = 0; i < 8; ++i) {
        int row = m0 + (half << 3) + i;
        outz[(size_t)row * N + col] = acc[t][i];
      }
    }
  }
}

// ---------------------------------------------------------------------------
// ConvLSTM gate math (Keras order i,f,c,o; activation=relu, rec=hard_sigmoid)
// ---------------------------------------------------------------------------
__global__ __launch_bounds__(256) void lstm_gate_kernel(
    const float* __restrict__ z, float* __restrict__ c,
    unsigned short* __restrict__ h, unsigned short* __restrict__ hseq,
    float* __restrict__ hf32, int HW, int C) {
  long long idx = (long long)blockIdx.x * blockDim.x + threadIdx.x;
  long long n = (long long)HW * C;
  if (idx >= n) return;
  int p  = (int)(idx / C);
  int ch = (int)(idx - (long long)p * C);
  const float* zp = z + (size_t)p * 4 * C;
  float ig = hsig(zp[ch]);
  float fg = hsig(zp[C + ch]);
  float cc = fmaxf(zp[2 * C + ch], 0.0f);
  float og = hsig(zp[3 * C + ch]);
  float cn = fg * c[idx] + ig * cc;
  float hn = og * fmaxf(cn, 0.0f);
  c[idx] = cn;
  unsigned short hb = f2bf(hn);
  h[idx] = hb;
  hseq[idx] = hb;
  if (hf32 != nullptr) hf32[idx] = hn;
}

// ---------------------------------------------------------------------------
// Elementwise helpers
// ---------------------------------------------------------------------------
__global__ __launch_bounds__(256) void cvt_f32_bf16_kernel(
    const float* __restrict__ in, unsigned short* __restrict__ out, long long n) {
  long long i = (long long)blockIdx.x * blockDim.x + threadIdx.x;
  if (i < n) out[i] = f2bf(in[i]);
}

__global__ __launch_bounds__(256) void maxpool_kernel(
    const unsigned short* __restrict__ in, unsigned short* __restrict__ out,
    int T, int H, int W, int C) {  // in dims; out is [T,H/2,W/2,C]
  int Ho = H >> 1, Wo = W >> 1;
  long long total = (long long)T * Ho * Wo * C;
  long long idx = (long long)blockIdx.x * blockDim.x + threadIdx.x;
  if (idx >= total) return;
  int c = (int)(idx % C); long long r = idx / C;
  int x = (int)(r % Wo); r /= Wo;
  int y = (int)(r % Ho); int t = (int)(r / Ho);
  const unsigned short* b =
      in + (((size_t)t * H + 2 * y) * W + 2 * x) * C + c;
  float m = bf2f(b[0]);
  m = fmaxf(m, bf2f(b[C]));
  m = fmaxf(m, bf2f(b[(size_t)W * C]));
  m = fmaxf(m, bf2f(b[(size_t)W * C + C]));
  out[idx] = f2bf(m);
}

__global__ __launch_bounds__(256) void concat_copy_kernel(
    const unsigned short* __restrict__ in, unsigned short* __restrict__ out,
    int T, int HW, int Cin, int Ctot, int coff) {
  long long total = (long long)T * HW * Cin;
  long long idx = (long long)blockIdx.x * blockDim.x + threadIdx.x;
  if (idx >= total) return;
  int c = (int)(idx % Cin);
  long long p = idx / Cin;  // t*HW + pixel
  out[(size_t)p * Ctot + coff + c] = in[idx];
}

__global__ __launch_bounds__(256) void upsample_concat_kernel(
    const unsigned short* __restrict__ in, unsigned short* __restrict__ out,
    int T, int Ho, int Wo, int Cin, int Ctot, int coff) {
  // in dims [T, Ho/2, Wo/2, Cin] -> out [T, Ho, Wo, Ctot] at channel coff
  long long total = (long long)T * Ho * Wo * Cin;
  long long idx = (long long)blockIdx.x * blockDim.x + threadIdx.x;
  if (idx >= total) return;
  int c = (int)(idx % Cin); long long r = idx / Cin;
  int x = (int)(r % Wo); r /= Wo;
  int y = (int)(r % Ho); int t = (int)(r / Ho);
  int Hi = Ho >> 1, Wi = Wo >> 1;
  unsigned short v = in[(((size_t)t * Hi + (y >> 1)) * Wi + (x >> 1)) * Cin + c];
  out[(((size_t)t * Ho + y) * Wo + x) * Ctot + coff + c] = v;
}

// ---------------------------------------------------------------------------
// Host-side helpers
// ---------------------------------------------------------------------------
static int pick_nt(int N) { return (N % 64 == 0) ? 4 : 1; }

static void repack(hipStream_t s, const float* w, unsigned short* wrep,
                   int K, int N) {
  int NT = pick_nt(N);
  int nKt = (K + 31) / 32;
  int nNg = (N + 16 * NT - 1) / (16 * NT);
  long long total = (long long)nNg * nKt * NT * 512;
  repack_weights_kernel<<<(unsigned)((total + 255) / 256), 256, 0, s>>>(
      w, wrep, K, N, nKt, NT, nNg);
}

static void launch_conv(hipStream_t s, const unsigned short* in,
                        const unsigned short* wrep, const float* bias,
                        float* out, int H, int W, int Cin, int N, int KH,
                        int accumulate, int Tbatch, size_t tin, size_t tout) {
  const int HW = H * W;
  const int nKt = (KH * KH * Cin + 31) / 32;
  const int NT = pick_nt(N);
  dim3 grid((unsigned)(HW / 64), (unsigned)((N + 16 * NT - 1) / (16 * NT)),
            (unsigned)Tbatch);
  if (NT == 4)
    conv_wmma_kernel2<4><<<grid, 128, 0, s>>>(in, wrep, bias, out, H, W, Cin,
                                              N, KH, nKt, accumulate, tin, tout);
  else
    conv_wmma_kernel2<1><<<grid, 128, 0, s>>>(in, wrep, bias, out, H, W, Cin,
                                              N, KH, nKt, accumulate, tin, tout);
}

static void run_layer(hipStream_t s, const unsigned short* in_seq,
                      unsigned short* out_seq, const unsigned short* wrep,
                      const unsigned short* urep, const float* bias,
                      int H, int W, int Cin, int C, int KH,
                      float* zall, unsigned short* hbuf, float* cbuf,
                      float* final_f32) {
  const int HW = H * W, N = 4 * C;
  // batched input-to-gate conv for all 4 timesteps at once
  launch_conv(s, in_seq, wrep, bias, zall, H, W, Cin, N, KH, 0,
              4, (size_t)HW * Cin, (size_t)HW * N);
  hipMemsetAsync(hbuf, 0, (size_t)HW * C * sizeof(unsigned short), s);
  hipMemsetAsync(cbuf, 0, (size_t)HW * C * sizeof(float), s);
  for (int t = 0; t < 4; ++t) {
    float* zt = zall + (size_t)t * HW * N;
    launch_conv(s, hbuf, urep, nullptr, zt, H, W, C, N, KH, 1, 1, 0, 0);
    long long n = (long long)HW * C;
    lstm_gate_kernel<<<(unsigned)((n + 255) / 256), 256, 0, s>>>(
        zt, cbuf, hbuf, out_seq + (size_t)t * HW * C,
        (t == 3) ? final_f32 : nullptr, HW, C);
  }
}

extern "C" void kernel_launch(void* const* d_in, const int* in_sizes, int n_in,
                              void* d_out, int out_size, void* d_ws, size_t ws_size,
                              hipStream_t stream) {
  (void)in_sizes; (void)n_in; (void)out_size; (void)ws_size;
  const int T = 4;
  struct Spec { int Cin, C, KH; };
  const Spec sp[15] = {
      {1, 32, 3},   {32, 32, 3},  {32, 64, 3},  {64, 64, 3},  {64, 128, 3},
      {128, 128, 3},{128, 256, 3},{256, 256, 3},{384, 128, 3},{128, 128, 3},
      {192, 64, 3}, {64, 64, 3},  {96, 32, 3},  {32, 32, 3},  {32, 1, 1}};

  const float* x = (const float*)d_in[0];
  const float* Wf[15]; const float* Uf[15]; const float* Bf[15];
  for (int i = 0; i < 15; ++i) {
    Wf[i] = (const float*)d_in[1 + 3 * i];
    Uf[i] = (const float*)d_in[2 + 3 * i];
    Bf[i] = (const float*)d_in[3 + 3 * i];
  }

  char* base = (char*)d_ws;
  size_t off = 0;
  auto carve = [&](size_t bytes) -> void* {
    void* p = base + off;
    off += (bytes + 255) & ~(size_t)255;
    return p;
  };

  auto rep_elems = [&](int K, int N) -> size_t {
    int NT = pick_nt(N);
    int nKt = (K + 31) / 32;
    int nNg = (N + 16 * NT - 1) / (16 * NT);
    return (size_t)nNg * nKt * NT * 512;
  };

  unsigned short* Wr[15]; unsigned short* Ur[15];
  for (int i = 0; i < 15; ++i) {
    int N = 4 * sp[i].C;
    int Kw = sp[i].KH * sp[i].KH * sp[i].Cin;
    int Ku = sp[i].KH * sp[i].KH * sp[i].C;
    Wr[i] = (unsigned short*)carve(rep_elems(Kw, N) * 2);
    Ur[i] = (unsigned short*)carve(rep_elems(Ku, N) * 2);
  }

  unsigned short* xb  = (unsigned short*)carve((size_t)T * 65536 * 1 * 2);
  unsigned short* s1a = (unsigned short*)carve((size_t)T * 65536 * 32 * 2);
  unsigned short* c1  = (unsigned short*)carve((size_t)T * 65536 * 32 * 2);
  unsigned short* p1  = (unsigned short*)carve((size_t)T * 16384 * 32 * 2);
  unsigned short* s2a = (unsigned short*)carve((size_t)T * 16384 * 64 * 2);
  unsigned short* c2  = (unsigned short*)carve((size_t)T * 16384 * 64 * 2);
  unsigned short* p2  = (unsigned short*)carve((size_t)T * 4096 * 64 * 2);
  unsigned short* s3a = (unsigned short*)carve((size_t)T * 4096 * 128 * 2);
  unsigned short* c3  = (unsigned short*)carve((size_t)T * 4096 * 128 * 2);
  unsigned short* p3  = (unsigned short*)carve((size_t)T * 1024 * 128 * 2);
  unsigned short* s4a = (unsigned short*)carve((size_t)T * 1024 * 256 * 2);
  unsigned short* c4  = (unsigned short*)carve((size_t)T * 1024 * 256 * 2);
  unsigned short* m5  = (unsigned short*)carve((size_t)T * 4096 * 384 * 2);
  unsigned short* s5a = (unsigned short*)carve((size_t)T * 4096 * 128 * 2);
  unsigned short* c5  = (unsigned short*)carve((size_t)T * 4096 * 128 * 2);
  unsigned short* m6  = (unsigned short*)carve((size_t)T * 16384 * 192 * 2);
  unsigned short* s6a = (unsigned short*)carve((size_t)T * 16384 * 64 * 2);
  unsigned short* c6  = (unsigned short*)carve((size_t)T * 16384 * 64 * 2);
  unsigned short* m7  = (unsigned short*)carve((size_t)T * 65536 * 96 * 2);
  unsigned short* s7a = (unsigned short*)carve((size_t)T * 65536 * 32 * 2);
  unsigned short* c7  = (unsigned short*)carve((size_t)T * 65536 * 32 * 2);
  unsigned short* oseq= (unsigned short*)carve((size_t)T * 65536 * 1 * 2);
  float* zall         = (float*)carve((size_t)T * 65536 * 128 * 4);
  unsigned short* hbuf= (unsigned short*)carve((size_t)65536 * 32 * 2);
  float* cbuf         = (float*)carve((size_t)65536 * 32 * 4);

  for (int i = 0; i < 15; ++i) {
    repack(stream, Wf[i], Wr[i], sp[i].KH * sp[i].KH * sp[i].Cin, 4 * sp[i].C);
    repack(stream, Uf[i], Ur[i], sp[i].KH * sp[i].KH * sp[i].C, 4 * sp[i].C);
  }
  {
    long long n = (long long)T * 65536;
    cvt_f32_bf16_kernel<<<(unsigned)((n + 255) / 256), 256, 0, stream>>>(x, xb, n);
  }

  // ---- encoder ----
  run_layer(stream, xb,  s1a, Wr[0], Ur[0], Bf[0], 256, 256, 1,  32, 3, zall, hbuf, cbuf, nullptr);
  run_layer(stream, s1a, c1,  Wr[1], Ur[1], Bf[1], 256, 256, 32, 32, 3, zall, hbuf, cbuf, nullptr);
  { long long n = (long long)T * 128 * 128 * 32;
    maxpool_kernel<<<(unsigned)((n + 255) / 256), 256, 0, stream>>>(c1, p1, T, 256, 256, 32); }
  run_layer(stream, p1,  s2a, Wr[2], Ur[2], Bf[2], 128, 128, 32, 64, 3, zall, hbuf, cbuf, nullptr);
  run_layer(stream, s2a, c2,  Wr[3], Ur[3], Bf[3], 128, 128, 64, 64, 3, zall, hbuf, cbuf, nullptr);
  { long long n = (long long)T * 64 * 64 * 64;
    maxpool_kernel<<<(unsigned)((n + 255) / 256), 256, 0, stream>>>(c2, p2, T, 128, 128, 64); }
  run_layer(stream, p2,  s3a, Wr[4], Ur[4], Bf[4], 64, 64, 64, 128, 3, zall, hbuf, cbuf, nullptr);
  run_layer(stream, s3a, c3,  Wr[5], Ur[5], Bf[5], 64, 64, 128, 128, 3, zall, hbuf, cbuf, nullptr);
  { long long n = (long long)T * 32 * 32 * 128;
    maxpool_kernel<<<(unsigned)((n + 255) / 256), 256, 0, stream>>>(c3, p3, T, 64, 64, 128); }
  run_layer(stream, p3,  s4a, Wr[6], Ur[6], Bf[6], 32, 32, 128, 256, 3, zall, hbuf, cbuf, nullptr);
  run_layer(stream, s4a, c4,  Wr[7], Ur[7], Bf[7], 32, 32, 256, 256, 3, zall, hbuf, cbuf, nullptr);

  // ---- decoder ----
  { long long n = (long long)T * 4096 * 128;
    concat_copy_kernel<<<(unsigned)((n + 255) / 256), 256, 0, stream>>>(c3, m5, T, 4096, 128, 384, 0); }
  { long long n = (long long)T * 64 * 64 * 256;
    upsample_concat_kernel<<<(unsigned)((n + 255) / 256), 256, 0, stream>>>(c4, m5, T, 64, 64, 256, 384, 128); }
  run_layer(stream, m5,  s5a, Wr[8], Ur[8], Bf[8], 64, 64, 384, 128, 3, zall, hbuf, cbuf, nullptr);
  run_layer(stream, s5a, c5,  Wr[9], Ur[9], Bf[9], 64, 64, 128, 128, 3, zall, hbuf, cbuf, nullptr);

  { long long n = (long long)T * 16384 * 64;
    concat_copy_kernel<<<(unsigned)((n + 255) / 256), 256, 0, stream>>>(c2, m6, T, 16384, 64, 192, 0); }
  { long long n = (long long)T * 128 * 128 * 128;
    upsample_concat_kernel<<<(unsigned)((n + 255) / 256), 256, 0, stream>>>(c5, m6, T, 128, 128, 128, 192, 64); }
  run_layer(stream, m6,  s6a, Wr[10], Ur[10], Bf[10], 128, 128, 192, 64, 3, zall, hbuf, cbuf, nullptr);
  run_layer(stream, s6a, c6,  Wr[11], Ur[11], Bf[11], 128, 128, 64, 64, 3, zall, hbuf, cbuf, nullptr);

  { long long n = (long long)T * 65536 * 32;
    concat_copy_kernel<<<(unsigned)((n + 255) / 256), 256, 0, stream>>>(c1, m7, T, 65536, 32, 96, 0); }
  { long long n = (long long)T * 256 * 256 * 64;
    upsample_concat_kernel<<<(unsigned)((n + 255) / 256), 256, 0, stream>>>(c6, m7, T, 256, 256, 64, 96, 32); }
  run_layer(stream, m7,  s7a, Wr[12], Ur[12], Bf[12], 256, 256, 96, 32, 3, zall, hbuf, cbuf, nullptr);
  run_layer(stream, s7a, c7,  Wr[13], Ur[13], Bf[13], 256, 256, 32, 32, 3, zall, hbuf, cbuf, nullptr);

  // ---- output ConvLSTM (1x1, C=1, return last h as f32 into d_out) ----
  run_layer(stream, c7, oseq, Wr[14], Ur[14], Bf[14], 256, 256, 32, 1, 1,
            zall, hbuf, cbuf, (float*)d_out);
}
